// CA_5523327942720
// MI455X (gfx1250) — compile-verified, matrix operand order checked by reading, not run on hardware
//
#include <hip/hip_runtime.h>

// ---------------------------------------------------------------------------
// Channel attention (CAM) block for MI455X / gfx1250, wave32 + WMMA f16.
//   energy = x1_d @ x2_d^T   (B=16, M=N=512, K=4096)   -> attention region
//   attention = softmax(energy, axis=-1)               (in place)
//   AV = attention^T @ x1_d  (M=512, N=4096, K=512)    -> out[:, 512:1024]
//   out[:, 0:512] = x1
// d_out = [ out : 16*1024*64*64 floats | attention : 16*512*512 floats ]
//
// GEMM structure: 128x64 tile per 256-thread workgroup; 8 waves (4x2), each
// wave a 32x32 register tile = 2 A-frags x 2 B-frags -> 4 WMMAs per K-step.
// Double-buffered LDS (one barrier per K-step); global loads for tile k+1
// issue before the WMMAs on tile k so LOADcnt overlaps the matrix work.
// ---------------------------------------------------------------------------

typedef __attribute__((ext_vector_type(16))) _Float16 v16h;
typedef __attribute__((ext_vector_type(8)))  _Float16 h8;
typedef __attribute__((ext_vector_type(8)))  float    v8f;

#define BATCH 16
#define CDIM  512
#define NPIX  4096
#define LDA   40   // halves per LDS tile row (32 + 8 pad); pitch 80 B (16B aligned)

__device__ __forceinline__ v8f wmma_f16(v16h a, v16h b, v8f c) {
  // (neg_a, A, neg_b, B, c_mod, C, reuse_a, reuse_b)
  return __builtin_amdgcn_wmma_f32_16x16x32_f16(false, a, false, b, (short)0, c,
                                                false, false);
}

// Build one 16x32 f16 fragment from an LDS tile stored row-major
// [row][k0..k31] (LDA halves pitch). CDNA5 16-bit A/B layout: lanes 0..15
// (row = lane) hold K={0..7,16..23}; lanes 16..31 hold K={8..15,24..31};
// VGPR q holds the K pair (q<4 ? 2q : 2q+8) + 8*hi.  Halves 0..7 and 16..23
// are contiguous, so the compiler fuses these into two ds_load_b128.
__device__ __forceinline__ v16h load_frag(const _Float16* tile, int row, int hi) {
  union { v16h v; unsigned int u[8]; } f;
  const unsigned int* p = (const unsigned int*)(tile + row * LDA);
  #pragma unroll
  for (int q = 0; q < 8; ++q) {
    const int k0 = ((q < 4) ? (2 * q) : (2 * q + 8)) + 8 * hi;  // even
    f.u[q] = p[k0 >> 1];
  }
  return f.v;
}

__device__ __forceinline__ h8 cvt8(const float4& lo, const float4& hi) {
  return h8{(_Float16)lo.x, (_Float16)lo.y, (_Float16)lo.z, (_Float16)lo.w,
            (_Float16)hi.x, (_Float16)hi.y, (_Float16)hi.z, (_Float16)hi.w};
}

// ---------------------------------------------------------------------------
// Kernel 1: copy x1 into out[:, 0:512, :, :]   (float4 coalesced)
// ---------------------------------------------------------------------------
__global__ __launch_bounds__(256) void k_copy_x1(const float4* __restrict__ x1,
                                                 float4* __restrict__ out) {
  const size_t g = (size_t)blockIdx.x * 256 + threadIdx.x;
  const size_t per = (size_t)CDIM * NPIX / 4;        // float4 per batch
  const size_t total = (size_t)BATCH * per;
  if (g >= total) return;
  const size_t b = g / per, rem = g % per;
  out[b * ((size_t)2 * CDIM * NPIX / 4) + rem] = x1[g];
}

// ---------------------------------------------------------------------------
// Kernel 2: energy[b,i,j] = sum_k x1[b,i,k] * x2[b,j,k]
// Tile: M=128 (i), N=64 (j), K-step 32. Double-buffered LDS.
// ---------------------------------------------------------------------------
__global__ __launch_bounds__(256) void k_energy(const float* __restrict__ x1,
                                                const float* __restrict__ x2,
                                                float* __restrict__ energy) {
  __shared__ __align__(16) _Float16 As[2][128 * LDA];  // x1 rows [i][k]
  __shared__ __align__(16) _Float16 Bs[2][64 * LDA];   // x2 rows [j][k]

  const int b  = blockIdx.z;
  const int i0 = blockIdx.x * 128;         // M tile (C1)
  const int j0 = blockIdx.y * 64;          // N tile (C2)
  const int t    = threadIdx.x;
  const int lane = t & 31;
  const int wave = t >> 5;
  const int wr = wave & 3;                 // M sub-tile: wr*32
  const int wc = wave >> 2;                // N sub-tile: wc*32
  const int hi   = lane >> 4;
  const int lrow = lane & 15;

  // staging maps
  const int ra = t >> 1, ca = (t & 1) * 16;   // A: 128 rows x 32 (16 floats/thread)
  const int rb = t >> 2, cb = (t & 3) * 8;    // B:  64 rows x 32 ( 8 floats/thread)
  const float* a_src = x1 + ((size_t)b * CDIM + i0 + ra) * NPIX + ca;
  const float* b_src = x2 + ((size_t)b * CDIM + j0 + rb) * NPIX + cb;
  const int aoff = ra * LDA + ca;
  const int boff = rb * LDA + cb;

  // prologue: stage tile 0 into buffer 0
  {
    const float4 a0 = *(const float4*)(a_src);
    const float4 a1 = *(const float4*)(a_src + 4);
    const float4 a2 = *(const float4*)(a_src + 8);
    const float4 a3 = *(const float4*)(a_src + 12);
    const float4 b0 = *(const float4*)(b_src);
    const float4 b1 = *(const float4*)(b_src + 4);
    *(h8*)&As[0][aoff]     = cvt8(a0, a1);
    *(h8*)&As[0][aoff + 8] = cvt8(a2, a3);
    *(h8*)&Bs[0][boff]     = cvt8(b0, b1);
  }
  __syncthreads();

  v8f acc00 = {}, acc01 = {}, acc10 = {}, acc11 = {};

  #pragma unroll 2
  for (int kk = 0; kk < NPIX; kk += 32) {
    const int cur = (kk >> 5) & 1, nxt = cur ^ 1;
    const bool has = (kk + 32) < NPIX;

    // issue next tile's global loads before the matrix work (overlap LOADcnt)
    float4 a0, a1, a2, a3, b0, b1;
    if (has) {
      a0 = *(const float4*)(a_src + kk + 32);
      a1 = *(const float4*)(a_src + kk + 36);
      a2 = *(const float4*)(a_src + kk + 40);
      a3 = *(const float4*)(a_src + kk + 44);
      b0 = *(const float4*)(b_src + kk + 32);
      b1 = *(const float4*)(b_src + kk + 36);
      if (kk + 64 < NPIX) {
        __builtin_prefetch(a_src + kk + 64, 0, 1);
        __builtin_prefetch(b_src + kk + 64, 0, 1);
      }
    }

    const v16h afA = load_frag(&As[cur][0], wr * 32 + lrow, hi);
    const v16h afB = load_frag(&As[cur][0], wr * 32 + 16 + lrow, hi);
    const v16h bf0 = load_frag(&Bs[cur][0], wc * 32 + lrow, hi);
    const v16h bf1 = load_frag(&Bs[cur][0], wc * 32 + 16 + lrow, hi);
    acc00 = wmma_f16(afA, bf0, acc00);
    acc01 = wmma_f16(afA, bf1, acc01);
    acc10 = wmma_f16(afB, bf0, acc10);
    acc11 = wmma_f16(afB, bf1, acc11);

    if (has) {
      *(h8*)&As[nxt][aoff]     = cvt8(a0, a1);
      *(h8*)&As[nxt][aoff + 8] = cvt8(a2, a3);
      *(h8*)&Bs[nxt][boff]     = cvt8(b0, b1);
    }
    __syncthreads();
  }

  // C/D layout: VGPR r, lane l -> M = r + 8*(l>=16), N = l%16
  float* ep = energy + ((size_t)b * CDIM + i0 + wr * 32) * CDIM + j0 + wc * 32;
  #pragma unroll
  for (int r = 0; r < 8; ++r) {
    const size_t row0 = (size_t)(8 * hi + r) * CDIM;
    const size_t row1 = (size_t)(16 + 8 * hi + r) * CDIM;
    ep[row0 + lrow]      = acc00[r];
    ep[row0 + 16 + lrow] = acc01[r];
    ep[row1 + lrow]      = acc10[r];
    ep[row1 + 16 + lrow] = acc11[r];
  }
}

// ---------------------------------------------------------------------------
// Kernel 3: in-place row softmax over last dim (512) of attention region.
// ---------------------------------------------------------------------------
__global__ __launch_bounds__(256) void k_softmax(float* __restrict__ attn) {
  __shared__ float red[256];
  float* p = attn + (size_t)blockIdx.x * CDIM;
  const int t = threadIdx.x;
  const float v0 = p[t], v1 = p[t + 256];

  red[t] = fmaxf(v0, v1);
  __syncthreads();
  #pragma unroll
  for (int s = 128; s > 0; s >>= 1) {
    if (t < s) red[t] = fmaxf(red[t], red[t + s]);
    __syncthreads();
  }
  const float m = red[0];
  __syncthreads();

  const float e0 = __expf(v0 - m), e1 = __expf(v1 - m);
  red[t] = e0 + e1;
  __syncthreads();
  #pragma unroll
  for (int s = 128; s > 0; s >>= 1) {
    if (t < s) red[t] += red[t + s];
    __syncthreads();
  }
  const float inv = 1.0f / red[0];
  p[t]       = e0 * inv;
  p[t + 256] = e1 * inv;
}

// ---------------------------------------------------------------------------
// Kernel 4: AV[b,j,n] = sum_i attn[b,i,j] * x1[b,i,n]  -> out[:, 512+j, n]
// A = attn^T, B = x1; both staged transposed ([j][i] / [n][i]) so fragments
// load with the same per-lane layout helper. M=128 (j), N=64 (n), K-step 32.
// ---------------------------------------------------------------------------
__global__ __launch_bounds__(256) void k_attnval(const float* __restrict__ attn,
                                                 const float* __restrict__ x1,
                                                 float* __restrict__ out) {
  __shared__ __align__(16) _Float16 As[2][128 * LDA];  // A[j][i]
  __shared__ __align__(16) _Float16 Bs[2][64 * LDA];   // B[n][i]

  const int b  = blockIdx.z;
  const int n0 = blockIdx.x * 64;          // pixel tile (N)
  const int j0 = blockIdx.y * 128;         // output-channel tile (M)
  const int t    = threadIdx.x;
  const int lane = t & 31;
  const int wave = t >> 5;
  const int wr = wave & 3;                 // j sub-tile: wr*32
  const int wc = wave >> 2;                // n sub-tile: wc*32
  const int hi   = lane >> 4;
  const int lrow = lane & 15;

  // staging: 32 K-rows; A covers 128 j-cols (16/thread), B 64 n-cols (8/thread)
  const int ii = t >> 3;                   // 0..31 (K index within tile)
  const int jq = (t & 7) * 16;             // 0..112
  const int nq = (t & 7) * 8;              // 0..56
  const float* a_base = attn + ((size_t)b * CDIM + ii) * CDIM + j0 + jq;
  const float* b_base = x1   + ((size_t)b * CDIM + ii) * NPIX + n0 + nq;

  // prologue: stage tile 0 into buffer 0
  {
    const float4 a0 = *(const float4*)(a_base);
    const float4 a1 = *(const float4*)(a_base + 4);
    const float4 a2 = *(const float4*)(a_base + 8);
    const float4 a3 = *(const float4*)(a_base + 12);
    const float4 b0 = *(const float4*)(b_base);
    const float4 b1 = *(const float4*)(b_base + 4);
    const float av[16] = {a0.x, a0.y, a0.z, a0.w, a1.x, a1.y, a1.z, a1.w,
                          a2.x, a2.y, a2.z, a2.w, a3.x, a3.y, a3.z, a3.w};
    const float bv[8]  = {b0.x, b0.y, b0.z, b0.w, b1.x, b1.y, b1.z, b1.w};
    #pragma unroll
    for (int s = 0; s < 16; ++s) As[0][(jq + s) * LDA + ii] = (_Float16)av[s];
    #pragma unroll
    for (int s = 0; s < 8; ++s)  Bs[0][(nq + s) * LDA + ii] = (_Float16)bv[s];
  }
  __syncthreads();

  v8f acc00 = {}, acc01 = {}, acc10 = {}, acc11 = {};

  #pragma unroll 2
  for (int kk = 0; kk < CDIM; kk += 32) {
    const int cur = (kk >> 5) & 1, nxt = cur ^ 1;
    const bool has = (kk + 32) < CDIM;

    float4 a0, a1, a2, a3, b0, b1;
    if (has) {
      const float* ap = a_base + (size_t)(kk + 32) * CDIM;
      const float* bp = b_base + (size_t)(kk + 32) * NPIX;
      a0 = *(const float4*)(ap);
      a1 = *(const float4*)(ap + 4);
      a2 = *(const float4*)(ap + 8);
      a3 = *(const float4*)(ap + 12);
      b0 = *(const float4*)(bp);
      b1 = *(const float4*)(bp + 4);
    }

    const v16h afA = load_frag(&As[cur][0], wr * 32 + lrow, hi);
    const v16h afB = load_frag(&As[cur][0], wr * 32 + 16 + lrow, hi);
    const v16h bf0 = load_frag(&Bs[cur][0], wc * 32 + lrow, hi);
    const v16h bf1 = load_frag(&Bs[cur][0], wc * 32 + 16 + lrow, hi);
    acc00 = wmma_f16(afA, bf0, acc00);
    acc01 = wmma_f16(afA, bf1, acc01);
    acc10 = wmma_f16(afB, bf0, acc10);
    acc11 = wmma_f16(afB, bf1, acc11);

    if (has) {
      const float av[16] = {a0.x, a0.y, a0.z, a0.w, a1.x, a1.y, a1.z, a1.w,
                            a2.x, a2.y, a2.z, a2.w, a3.x, a3.y, a3.z, a3.w};
      const float bv[8]  = {b0.x, b0.y, b0.z, b0.w, b1.x, b1.y, b1.z, b1.w};
      #pragma unroll
      for (int s = 0; s < 16; ++s) As[nxt][(jq + s) * LDA + ii] = (_Float16)av[s];
      #pragma unroll
      for (int s = 0; s < 8; ++s)  Bs[nxt][(nq + s) * LDA + ii] = (_Float16)bv[s];
    }
    __syncthreads();
  }

  // out channel = 512 + j0 + wr*32 + {0,16} + (8*hi + r); pixel = n0 + wc*32 + {0,16} + lrow
  float* op = out + ((size_t)b * (2 * CDIM) + CDIM + j0 + wr * 32) * NPIX
                  + n0 + wc * 32;
  #pragma unroll
  for (int r = 0; r < 8; ++r) {
    const size_t row0 = (size_t)(8 * hi + r) * NPIX;
    const size_t row1 = (size_t)(16 + 8 * hi + r) * NPIX;
    op[row0 + lrow]      = acc00[r];
    op[row0 + 16 + lrow] = acc01[r];
    op[row1 + lrow]      = acc10[r];
    op[row1 + 16 + lrow] = acc11[r];
  }
}

// ---------------------------------------------------------------------------
extern "C" void kernel_launch(void* const* d_in, const int* in_sizes, int n_in,
                              void* d_out, int out_size, void* d_ws, size_t ws_size,
                              hipStream_t stream) {
  (void)in_sizes; (void)n_in; (void)out_size; (void)d_ws; (void)ws_size;
  const float* x1 = (const float*)d_in[0];
  const float* x2 = (const float*)d_in[1];
  float* out  = (float*)d_out;
  float* attn = out + (size_t)BATCH * 2 * CDIM * NPIX;   // 67,108,864 floats in

  // 1) out[:, 0:512] = x1
  {
    const size_t nvec4 = (size_t)BATCH * CDIM * NPIX / 4;
    const int blocks = (int)((nvec4 + 255) / 256);
    k_copy_x1<<<blocks, 256, 0, stream>>>((const float4*)x1, (float4*)out);
  }
  // 2) energy -> attention region
  k_energy<<<dim3(CDIM / 128, CDIM / 64, BATCH), 256, 0, stream>>>(x1, x2, attn);
  // 3) softmax rows in place
  k_softmax<<<BATCH * CDIM, 256, 0, stream>>>(attn);
  // 4) attention^T @ x1 -> out[:, 512:1024]
  k_attnval<<<dim3(NPIX / 64, CDIM / 128, BATCH), 256, 0, stream>>>(attn, x1, out);
}
